// TransformNetOld_7705171329413
// MI455X (gfx1250) — compile-verified, matrix operand order checked by reading, not run on hardware
//
#include <hip/hip_runtime.h>
#include <math.h>

typedef __attribute__((ext_vector_type(16))) __bf16  v16bf;
typedef __attribute__((ext_vector_type(8)))  float   v8f;
typedef __attribute__((ext_vector_type(2)))  float   v2f;

constexpr int   BB     = 8;
constexpr int   MPTS   = 1024;
constexpr int   KNN    = 20;
constexpr int   NPTS   = BB * MPTS;       // 8192 points
constexpr int   R2     = NPTS * KNN;      // 163840 edge rows
constexpr float EPSBN  = 1e-5f;
constexpr float SLOPE  = 0.2f;
constexpr float NEG_INF = -3.402823466e38f;

#define DEV __device__ __forceinline__

DEV float lrelu(float x) { return x >= 0.f ? x : SLOPE * x; }

DEV unsigned short f2bf(float f) {            // round-to-nearest-even f32 -> bf16
  unsigned u = __float_as_uint(f);
  u += 0x7FFFu + ((u >> 16) & 1u);
  return (unsigned short)(u >> 16);
}

union V2f  { float f[2]; v2f v; };
union V8f  { float f[8]; v8f v; };
union BF16x16 { uint4 q[2]; unsigned short u[16]; v16bf v; };

// ---------------------------------------------------------------- utilities
__global__ void k_zero(float* p, int n) {
  int i = blockIdx.x * blockDim.x + threadIdx.x;
  if (i < n) p[i] = 0.f;
}

// ------------------------------------------------- kNN #1: 3D squared dists
__global__ void k_dist3(const float* __restrict__ p, float* __restrict__ D) {
  int t = blockIdx.x * blockDim.x + threadIdx.x;
  if (t >= BB * MPTS * MPTS) return;
  int j = t % MPTS; int r = t / MPTS; int i = r % MPTS; int b = r / MPTS;
  const float* pi = p + (long long)(b * MPTS + i) * 3;
  const float* pj = p + (long long)(b * MPTS + j) * 3;
  float d0 = pi[0] - pj[0], d1 = pi[1] - pj[1], d2 = pi[2] - pj[2];
  D[t] = d0 * d0 + d1 * d1 + d2 * d2;
}

// ---------------------------- top-k (k=20) per row of a [8192,1024] matrix
// Row is staged into LDS with CDNA5 GLOBAL_LOAD_ASYNC_TO_LDS_B128 (GVS mode:
// uniform SGPR-pair base + per-lane byte offset), tracked by ASYNCcnt.
__global__ __launch_bounds__(32) void k_topk(const float* __restrict__ D,
                                             int* __restrict__ idx) {
  __shared__ float dv[MPTS];
  int row  = blockIdx.x;
  int lane = threadIdx.x;
  const float* Dr = D + (long long)row * MPTS;   // block-uniform -> SGPRs
  unsigned ldsbase = (unsigned)(uintptr_t)(void*)dv;   // LDS aperture: addr[31:0]
#pragma unroll
  for (int j4 = 0; j4 < MPTS; j4 += 128) {       // 32 lanes x 16B = 512B/iter
    unsigned goff   = (unsigned)(j4 + lane * 4) * 4u;  // byte offset in row
    unsigned ldsoff = ldsbase + goff;
    asm volatile("global_load_async_to_lds_b128 %0, %1, %2"
                 :: "v"(ldsoff), "v"(goff), "s"(Dr) : "memory");
  }
  asm volatile("s_wait_asynccnt 0" ::: "memory");
  __syncthreads();
  for (int t = 0; t < KNN; ++t) {
    float bv = 3.4e38f; int bi = MPTS;
    for (int j = lane; j < MPTS; j += 32) {
      float v = dv[j];
      if (v < bv || (v == bv && j < bi)) { bv = v; bi = j; }
    }
    for (int off = 16; off; off >>= 1) {
      float ov = __shfl_xor(bv, off, 32);
      int   oi = __shfl_xor(bi, off, 32);
      if (ov < bv || (ov == bv && oi < bi)) { bv = ov; bi = oi; }
    }
    if (lane == 0) { idx[row * KNN + t] = bi; dv[bi] = 3.4e38f; }
    __syncthreads();
  }
}

// ---------------------------------------- edge conv 1: h1 = [xi, xj-xi]@W1
__global__ void k_edge1(const float* __restrict__ p, const int* __restrict__ idx,
                        const float* __restrict__ W1, float* __restrict__ h1) {
  __shared__ float w[6 * 64];
  for (int t = threadIdx.x; t < 6 * 64; t += blockDim.x) w[t] = W1[t];
  __syncthreads();
  int r = blockIdx.x * blockDim.x + threadIdx.x;
  if (r >= R2) return;
  int pm = r / KNN;
  int b  = pm / MPTS;
  int j  = idx[r];
  const float* xi = p + (long long)pm * 3;
  const float* xj = p + (long long)(b * MPTS + j) * 3;
  float e[6] = { xi[0], xi[1], xi[2], xj[0]-xi[0], xj[1]-xi[1], xj[2]-xi[2] };
  float* out = h1 + (long long)r * 64;
  for (int c = 0; c < 64; ++c) {
    float s = 0.f;
#pragma unroll
    for (int d = 0; d < 6; ++d) s += e[d] * w[d * 64 + c];
    out[c] = s;
  }
}

// ------------------------------- per-channel sum / sumsq (C = 64 or 128)
template <int C>
__global__ void k_stats(const float* __restrict__ h, int R,
                        float* __restrict__ ssum, float* __restrict__ ssq) {
  __shared__ float ls[256], ls2[256];
  const int ROWS = 2048;
  int t = threadIdx.x;
  int c = t % C;
  int G = 256 / C;
  int r0   = blockIdx.x * ROWS + t / C;
  int rend = blockIdx.x * ROWS + ROWS; if (rend > R) rend = R;
  float s = 0.f, s2 = 0.f;
  for (int r = r0; r < rend; r += G) {
    float v = h[(long long)r * C + c];
    s += v; s2 += v * v;
  }
  ls[t] = s; ls2[t] = s2; __syncthreads();
  for (int off = 128; off >= C; off >>= 1) {
    if (t < off) { ls[t] += ls[t + off]; ls2[t] += ls2[t + off]; }
    __syncthreads();
  }
  if (t < C) { atomicAdd(&ssum[c], ls[t]); atomicAdd(&ssq[c], ls2[t]); }
}

__global__ void k_meanvar(const float* __restrict__ ssum, const float* __restrict__ ssq,
                          float Rinv, int C, float* __restrict__ mean,
                          float* __restrict__ istd) {
  int c = blockIdx.x * blockDim.x + threadIdx.x;
  if (c >= C) return;
  float m = ssum[c] * Rinv;
  float v = ssq[c] * Rinv - m * m;
  mean[c] = m;
  istd[c] = rsqrtf(v + EPSBN);
}

// ----------------------- h2 = lrelu(bn(h1)) @ W2  (64 -> 128, W2 in LDS)
__global__ __launch_bounds__(256) void k_bn_gemm2(
    const float* __restrict__ h1, const float* __restrict__ mean,
    const float* __restrict__ istd, const float* __restrict__ g,
    const float* __restrict__ bb, const float* __restrict__ W2,
    float* __restrict__ h2) {
  __shared__ float w[64 * 128];
  __shared__ float sm[64], si[64], sg[64], sb[64];
  for (int t = threadIdx.x; t < 64 * 128; t += blockDim.x) w[t] = W2[t];
  if (threadIdx.x < 64) {
    sm[threadIdx.x] = mean[threadIdx.x]; si[threadIdx.x] = istd[threadIdx.x];
    sg[threadIdx.x] = g[threadIdx.x];    sb[threadIdx.x] = bb[threadIdx.x];
  }
  __syncthreads();
  int r = blockIdx.x * blockDim.x + threadIdx.x;
  if (r >= R2) return;
  float a[64];
  const float* hr = h1 + (long long)r * 64;
#pragma unroll
  for (int d = 0; d < 64; ++d)
    a[d] = lrelu((hr[d] - sm[d]) * si[d] * sg[d] + sb[d]);
  float* out = h2 + (long long)r * 128;
  for (int c = 0; c < 128; ++c) {
    float s = 0.f;
#pragma unroll
    for (int d = 0; d < 64; ++d) s += a[d] * w[d * 128 + c];
    out[c] = s;
  }
}

// --------------------- x1 = max_k lrelu(bn(h2))  -> [8192,128]
__global__ void k_bn_maxk(const float* __restrict__ h2, const float* __restrict__ mean,
                          const float* __restrict__ istd, const float* __restrict__ g,
                          const float* __restrict__ bb, float* __restrict__ x1) {
  int t = blockIdx.x * blockDim.x + threadIdx.x;
  if (t >= NPTS * 128) return;
  int c = t & 127, pt = t >> 7;
  float m = mean[c], is = istd[c], gg = g[c], bv = bb[c];
  float mx = NEG_INF;
  const float* base = h2 + (long long)pt * KNN * 128 + c;
  for (int k = 0; k < KNN; ++k)
    mx = fmaxf(mx, lrelu((base[k * 128] - m) * is * gg + bv));
  x1[t] = mx;
}

__global__ void k_sqnorm(const float* __restrict__ x1, float* __restrict__ sq) {
  int i = blockIdx.x * blockDim.x + threadIdx.x;
  if (i >= NPTS) return;
  float s = 0.f;
  const float* r = x1 + (long long)i * 128;
  for (int c = 0; c < 128; ++c) s += r[c] * r[c];
  sq[i] = s;
}

// ---------- kNN #2 gram via V_WMMA_F32_16X16X4_F32; fused dist epilogue
__global__ __launch_bounds__(256) void k_gram_wmma(
    const float* __restrict__ x1, const float* __restrict__ sq,
    float* __restrict__ D) {
  int wave = threadIdx.x >> 5;
  int lane = threadIdx.x & 31;
  int tile = blockIdx.x * 8 + wave;          // 8*64*64 tiles total
  int jt = tile & 63; tile >>= 6;
  int it = tile & 63; int b = tile >> 6;
  int lm = lane & 15;
  int khalf = (lane < 16) ? 0 : 2;
  const float* arow = x1 + ((long long)(b * MPTS + it * 16 + lm)) * 128;
  const float* brow = x1 + ((long long)(b * MPTS + jt * 16 + lm)) * 128;
  v8f acc = {};
  for (int k = 0; k < 128; k += 4) {
    V2f a, bf;
    a.f[0]  = arow[k + khalf]; a.f[1]  = arow[k + khalf + 1];
    bf.f[0] = brow[k + khalf]; bf.f[1] = brow[k + khalf + 1];
    acc = __builtin_amdgcn_wmma_f32_16x16x4_f32(false, a.v, false, bf.v,
                                                (short)0, acc, false, false);
  }
  V8f r; r.v = acc;
  int i0 = it * 16 + ((lane < 16) ? 0 : 8);
  int j  = jt * 16 + lm;
  float sj = sq[b * MPTS + j];
  float* Db = D + (long long)b * MPTS * MPTS;
#pragma unroll
  for (int r8 = 0; r8 < 8; ++r8) {
    int i = i0 + r8;
    Db[(long long)i * MPTS + j] = sq[b * MPTS + i] + sj - 2.f * r.f[r8];
  }
}

// ----------------------- W3 [256,1024] f32 -> bf16 column-major [1024,256]
__global__ void k_w3t(const float* __restrict__ W3, unsigned short* __restrict__ W3t) {
  int t = blockIdx.x * blockDim.x + threadIdx.x;
  if (t >= 1024 * 256) return;
  int k = t & 255, n = t >> 8;
  W3t[t] = f2bf(W3[(long long)k * 1024 + n]);
}

// ---- edge conv 2 GEMM: 256->1024 via V_WMMA_F32_16X16X32_BF16.
// One block = one point (32x256 bf16 A-tile, rows 20..31 padded) x 256 cols.
// pass 0: accumulate BN sums;  pass 1: bn+lrelu+max over the 20 neighbors.
__global__ __launch_bounds__(512) void k_edge2_wmma(
    const float* __restrict__ x1, const int* __restrict__ idx,
    const unsigned short* __restrict__ W3t,
    const float* __restrict__ mean, const float* __restrict__ istd,
    const float* __restrict__ g, const float* __restrict__ bb,
    float* __restrict__ ssum, float* __restrict__ ssq,
    float* __restrict__ x3, int pass) {
  __shared__ unsigned short A[32 * 256];     // 16 KB bf16 edge-feature tile
  int cb = blockIdx.x & 3;                   // column block (256 channels)
  int pm = blockIdx.x >> 2;                  // b*1024 + m
  int b  = pm / MPTS;
  for (int t = threadIdx.x; t < 32 * 256; t += 512) {
    int row = t >> 8, col = t & 255;
    float v = 0.f;
    if (row < KNN) {
      int j   = idx[pm * KNN + row];
      int pmj = b * MPTS + j;
      if (col < 128) v = x1[(long long)pm * 128 + col];
      else {
        int c = col - 128;
        v = x1[(long long)pmj * 128 + c] - x1[(long long)pm * 128 + c];
      }
    }
    A[t] = f2bf(v);
  }
  __syncthreads();
  int wave = threadIdx.x >> 5, lane = threadIdx.x & 31;
  int lm = lane & 15;
  int n  = cb * 256 + wave * 16 + lm;        // this lane's output channel
  int koff = (lane < 16) ? 0 : 8;            // 16-bit A/B fragment K pattern
  v8f acc0 = {}, acc1 = {};
  const unsigned short* Bp = W3t + (long long)n * 256;
  for (int ks = 0; ks < 8; ++ks) {           // K = 256 = 8 x 32
    int k0 = ks * 32 + koff;
    if (ks < 7) __builtin_prefetch(Bp + k0 + 32, 0, 1);  // next B fragment
    BF16x16 fa0, fa1, fb;
    fa0.q[0] = *(const uint4*)(A + lm * 256 + k0);
    fa0.q[1] = *(const uint4*)(A + lm * 256 + k0 + 16);
    fa1.q[0] = *(const uint4*)(A + (16 + lm) * 256 + k0);
    fa1.q[1] = *(const uint4*)(A + (16 + lm) * 256 + k0 + 16);
    fb.q[0]  = *(const uint4*)(Bp + k0);
    fb.q[1]  = *(const uint4*)(Bp + k0 + 16);
    acc0 = __builtin_amdgcn_wmma_f32_16x16x32_bf16(false, fa0.v, false, fb.v,
                                                   (short)0, acc0, false, false);
    acc1 = __builtin_amdgcn_wmma_f32_16x16x32_bf16(false, fa1.v, false, fb.v,
                                                   (short)0, acc1, false, false);
  }
  V8f r0, r1; r0.v = acc0; r1.v = acc1;
  int rowoff = (lane < 16) ? 0 : 8;
  if (pass == 0) {
    float s = 0.f, s2 = 0.f;
#pragma unroll
    for (int r8 = 0; r8 < 8; ++r8) {
      int row = r8 + rowoff;
      if (row      < KNN) { float v = r0.f[r8]; s += v; s2 += v * v; }
      if (row + 16 < KNN) { float v = r1.f[r8]; s += v; s2 += v * v; }
    }
    s  += __shfl_xor(s, 16, 32);             // lanes L / L+16 share channel n
    s2 += __shfl_xor(s2, 16, 32);
    if (lane < 16) { atomicAdd(&ssum[n], s); atomicAdd(&ssq[n], s2); }
  } else {
    float m = mean[n], is = istd[n], gg = g[n], bv = bb[n];
    float mx = NEG_INF;
#pragma unroll
    for (int r8 = 0; r8 < 8; ++r8) {
      int row = r8 + rowoff;
      if (row      < KNN) mx = fmaxf(mx, lrelu((r0.f[r8] - m) * is * gg + bv));
      if (row + 16 < KNN) mx = fmaxf(mx, lrelu((r1.f[r8] - m) * is * gg + bv));
    }
    mx = fmaxf(mx, __shfl_xor(mx, 16, 32));
    if (lane < 16) x3[(long long)pm * 1024 + n] = mx;
  }
}

// ------------------------------------------------- global max over points
__global__ void k_gmax(const float* __restrict__ x3, float* __restrict__ gfeat) {
  int t = blockIdx.x * blockDim.x + threadIdx.x;
  if (t >= BB * 1024) return;
  int c = t & 1023, b = t >> 10;
  float mx = NEG_INF;
  const float* base = x3 + (long long)b * MPTS * 1024 + c;
  for (int m = 0; m < MPTS; ++m) mx = fmaxf(mx, base[(long long)m * 1024]);
  gfeat[t] = mx;
}

// ------------- head: Linear-BN-Linear-BN-Linear on [8,1024] (single block)
__global__ __launch_bounds__(256) void k_head(
    const float* __restrict__ gfeat,
    const float* __restrict__ W4, const float* __restrict__ g4, const float* __restrict__ b4,
    const float* __restrict__ W5, const float* __restrict__ g5, const float* __restrict__ b5,
    const float* __restrict__ W6, const float* __restrict__ bias6,
    float* __restrict__ out) {
  __shared__ float h4[8 * 512];
  __shared__ float h5[8 * 256];
  int t = threadIdx.x;
  for (int o = t; o < 8 * 512; o += 256) {
    int row = o >> 9, col = o & 511;
    float s = 0.f;
    const float* gr = gfeat + row * 1024;
    for (int d = 0; d < 1024; ++d) s += gr[d] * W4[(long long)d * 512 + col];
    h4[o] = s;
  }
  __syncthreads();
  for (int c = t; c < 512; c += 256) {
    float s = 0.f, s2 = 0.f;
    for (int r = 0; r < 8; ++r) { float v = h4[r * 512 + c]; s += v; s2 += v * v; }
    float m = s * 0.125f, var = s2 * 0.125f - m * m;
    float is = rsqrtf(var + EPSBN);
    for (int r = 0; r < 8; ++r)
      h4[r * 512 + c] = g4[c] * (h4[r * 512 + c] - m) * is + b4[c];
  }
  __syncthreads();
  for (int o = t; o < 8 * 256; o += 256) {
    int row = o >> 8, col = o & 255;
    float s = 0.f;
    for (int d = 0; d < 512; ++d) s += h4[row * 512 + d] * W5[(long long)d * 256 + col];
    h5[o] = s;
  }
  __syncthreads();
  for (int c = t; c < 256; c += 256) {
    float s = 0.f, s2 = 0.f;
    for (int r = 0; r < 8; ++r) { float v = h5[r * 256 + c]; s += v; s2 += v * v; }
    float m = s * 0.125f, var = s2 * 0.125f - m * m;
    float is = rsqrtf(var + EPSBN);
    for (int r = 0; r < 8; ++r)
      h5[r * 256 + c] = g5[c] * (h5[r * 256 + c] - m) * is + b5[c];
  }
  __syncthreads();
  if (t < 8 * 9) {
    int row = t / 9, col = t % 9;
    float s = 0.f;
    for (int d = 0; d < 256; ++d) s += h5[row * 256 + d] * W6[d * 9 + col];
    out[t] = s + bias6[col];
  }
}

// --------------------------------------------------------------------------
extern "C" void kernel_launch(void* const* d_in, const int* in_sizes, int n_in,
                              void* d_out, int out_size, void* d_ws, size_t ws_size,
                              hipStream_t stream) {
  (void)in_sizes; (void)n_in; (void)out_size; (void)ws_size;
  const float* p    = (const float*)d_in[0];
  const float* W1   = (const float*)d_in[1];
  const float* g1   = (const float*)d_in[2];
  const float* b1   = (const float*)d_in[3];
  const float* W2   = (const float*)d_in[4];
  const float* g2   = (const float*)d_in[5];
  const float* b2   = (const float*)d_in[6];
  const float* W3   = (const float*)d_in[7];
  const float* g3   = (const float*)d_in[8];
  const float* b3   = (const float*)d_in[9];
  const float* W4   = (const float*)d_in[10];
  const float* g4   = (const float*)d_in[11];
  const float* b4   = (const float*)d_in[12];
  const float* W5   = (const float*)d_in[13];
  const float* g5   = (const float*)d_in[14];
  const float* b5   = (const float*)d_in[15];
  const float* W6   = (const float*)d_in[16];
  const float* bias6= (const float*)d_in[17];

  char* ws = (char*)d_ws;
  size_t off = 0;
  auto alloc = [&](size_t bytes) -> char* {
    char* r = ws + off;
    off = (off + bytes + 255) & ~(size_t)255;
    return r;
  };
  float* D    = (float*)alloc((size_t)BB * MPTS * MPTS * 4);   // 33.5 MB (reused for both kNNs)
  int*   idx1 = (int*)  alloc((size_t)R2 * 4);
  int*   idx2 = (int*)  alloc((size_t)R2 * 4);
  float* h1   = (float*)alloc((size_t)R2 * 64 * 4);            // 41.9 MB
  float* h2   = (float*)alloc((size_t)R2 * 128 * 4);           // 83.9 MB
  float* x1   = (float*)alloc((size_t)NPTS * 128 * 4);
  float* sq1  = (float*)alloc((size_t)NPTS * 4);
  float* x3   = (float*)alloc((size_t)NPTS * 1024 * 4);        // 33.5 MB
  unsigned short* W3t = (unsigned short*)alloc((size_t)1024 * 256 * 2);
  float* stats = (float*)alloc((size_t)4 * (64 + 128 + 1024) * 4);
  float* ss1 = stats;       float* sqs1 = ss1 + 64;   float* mn1 = sqs1 + 64;   float* is1 = mn1 + 64;
  float* ss2 = is1 + 64;    float* sqs2 = ss2 + 128;  float* mn2 = sqs2 + 128;  float* is2 = mn2 + 128;
  float* ss3 = is2 + 128;   float* sqs3 = ss3 + 1024; float* mn3 = sqs3 + 1024; float* is3 = mn3 + 1024;
  float* gfeat = (float*)alloc((size_t)BB * 1024 * 4);

  const int NSTAT = 4 * (64 + 128 + 1024);
  k_zero<<<(NSTAT + 255) / 256, 256, 0, stream>>>(stats, NSTAT);

  // --- DynamicEdgeConv 1
  int ND = BB * MPTS * MPTS;
  k_dist3<<<(ND + 255) / 256, 256, 0, stream>>>(p, D);
  k_topk<<<NPTS, 32, 0, stream>>>(D, idx1);
  k_edge1<<<(R2 + 255) / 256, 256, 0, stream>>>(p, idx1, W1, h1);
  k_stats<64><<<R2 / 2048, 256, 0, stream>>>(h1, R2, ss1, sqs1);
  k_meanvar<<<1, 64, 0, stream>>>(ss1, sqs1, 1.f / R2, 64, mn1, is1);
  k_bn_gemm2<<<(R2 + 255) / 256, 256, 0, stream>>>(h1, mn1, is1, g1, b1, W2, h2);
  k_stats<128><<<R2 / 2048, 256, 0, stream>>>(h2, R2, ss2, sqs2);
  k_meanvar<<<1, 128, 0, stream>>>(ss2, sqs2, 1.f / R2, 128, mn2, is2);
  k_bn_maxk<<<(NPTS * 128 + 255) / 256, 256, 0, stream>>>(h2, mn2, is2, g2, b2, x1);

  // --- DynamicEdgeConv 2
  k_sqnorm<<<(NPTS + 255) / 256, 256, 0, stream>>>(x1, sq1);
  k_gram_wmma<<<BB * 64 * 64 / 8, 256, 0, stream>>>(x1, sq1, D);
  k_topk<<<NPTS, 32, 0, stream>>>(D, idx2);
  k_w3t<<<(1024 * 256 + 255) / 256, 256, 0, stream>>>(W3, W3t);
  k_edge2_wmma<<<NPTS * 4, 512, 0, stream>>>(x1, idx2, W3t, mn3, is3, g3, b3,
                                             ss3, sqs3, x3, 0);
  k_meanvar<<<4, 256, 0, stream>>>(ss3, sqs3, 1.f / R2, 1024, mn3, is3);
  k_edge2_wmma<<<NPTS * 4, 512, 0, stream>>>(x1, idx2, W3t, mn3, is3, g3, b3,
                                             ss3, sqs3, x3, 1);

  // --- global pool + head
  k_gmax<<<(BB * 1024 + 255) / 256, 256, 0, stream>>>(x3, gfeat);
  k_head<<<1, 256, 0, stream>>>(gfeat, W4, g4, b4, W5, g5, b5, W6, bias6,
                                (float*)d_out);
}
